// MHA_14534169329952
// MI455X (gfx1250) — matrix operand order repaired and checked by reading, not compiled, not run on hardware
//
#include <hip/hip_runtime.h>

// ---------------------------------------------------------------------------
// CDNA5 (gfx1250) wave32 BF16 WMMA implementation of downsampled attention.
//   q = X Wq + bq ; kd = pool(K) Wk + bk ; vd = pool(V) Wv + bv
//   out = softmax(q kd^T / 8) vd  Wo + bo
// ---------------------------------------------------------------------------

typedef __attribute__((ext_vector_type(16))) __bf16 v16bf;
typedef __attribute__((ext_vector_type(8)))  __bf16 v8bf;
typedef __attribute__((ext_vector_type(8)))  float  v8f;

__device__ __forceinline__ v8f vzero8() {
    v8f z;
#pragma unroll
    for (int i = 0; i < 8; ++i) z[i] = 0.0f;
    return z;
}

__device__ __forceinline__ v16bf cat16(v8bf lo, v8bf hi) {
    return __builtin_shufflevector(lo, hi, 0,1,2,3,4,5,6,7,8,9,10,11,12,13,14,15);
}

__device__ __forceinline__ __bf16 to_bf16(float x)  { return (__bf16)x; }
__device__ __forceinline__ __bf16 to_bf16(__bf16 x) { return x; }

// A fragment (16x32 bf16, row-major LDS tile, row stride RS elements).
// Lane (g = lane>>4, m = lane&15): a[e] = A[m][koff + (e&7) + ((e&8)<<1) + 8g]
// -> two contiguous 16B segments at +8g and +16+8g (RS*2 multiple of 16 keeps align).
template<int RS>
__device__ __forceinline__ v16bf load_frag_a(const __bf16* base, int row, int koff, int g) {
    const __bf16* p = base + row * RS + koff;
    v8bf lo = *(const v8bf*)(p + 8 * g);
    v8bf hi = *(const v8bf*)(p + 16 + 8 * g);
    return cat16(lo, hi);
}

// B fragment (32x16 bf16) from K-contiguous-per-column LDS tile Bs[col][k]:
// lane (g, n = lane&15): b[e] = Bs[col][koff + (e&7) + ((e&8)<<1) + 8g]  -> two 16B segments.
template<int RS>
__device__ __forceinline__ v16bf load_frag_b(const __bf16* base, int col, int koff, int g) {
    const __bf16* p = base + col * RS + koff;
    v8bf lo = *(const v8bf*)(p + 8 * g);
    v8bf hi = *(const v8bf*)(p + 16 + 8 * g);
    return cat16(lo, hi);
}

__device__ __forceinline__ v8f wmma_bf16(v16bf a, v16bf b, v8f c) {
    return __builtin_amdgcn_wmma_f32_16x16x32_bf16(false, a, false, b, (short)0, c,
                                                   false, false);
}

// ---------------------------------------------------------------------------
// Kernel 1: mean-pool key/value along S into 64 buckets of 64.
// grid (64, B, 2), block 256. out: [B,64,1024] f32
// ---------------------------------------------------------------------------
__global__ __launch_bounds__(256) void pool_k(const float* __restrict__ key,
                                              const float* __restrict__ value,
                                              float* __restrict__ kp,
                                              float* __restrict__ vp) {
    const int t = threadIdx.x;
    const int p = blockIdx.x, b = blockIdx.y;
    const float* src = blockIdx.z ? value : key;
    float*       dst = blockIdx.z ? vp : kp;
    const float* base = src + ((size_t)(b * 4096 + p * 64)) * 1024;
    float acc[4] = {0.f, 0.f, 0.f, 0.f};
    for (int s = 0; s < 64; ++s) {
#pragma unroll
        for (int i = 0; i < 4; ++i)
            acc[i] += base[(size_t)s * 1024 + t + 256 * i];
    }
#pragma unroll
    for (int i = 0; i < 4; ++i)
        dst[(size_t)(b * 64 + p) * 1024 + t + 256 * i] = acc[i] * (1.0f / 64.0f);
}

// ---------------------------------------------------------------------------
// Kernel: WMMA GEMM  C[M][1024] = A[M][1024] * W[1024][1024] + bias[1024]
// A: f32 or bf16 (converted to bf16 while staging to LDS); W: f32 (converted).
// Block 256 threads = 8 waves in a 4x2 grid; block tile 128x128, BK=64.
// Each wave owns a 32x64 slab = 8 accumulator tiles -> 16 WMMA per K-chunk.
// grid (1024/128, M/128).
// ---------------------------------------------------------------------------
template<typename AT, bool OUT_BF16>
__global__ __launch_bounds__(256) void gemm_bias_k(const AT* __restrict__ A,
                                                   const float* __restrict__ W,
                                                   const float* __restrict__ bias,
                                                   void* __restrict__ outp, int M) {
    constexpr int Kd = 1024, Nd = 1024, BK = 64;
    __shared__ __bf16 As[128][72];   // row stride 144 B (16B aligned)
    __shared__ __bf16 Bs[128][72];   // Bs[col][k], K contiguous per column

    const int t = threadIdx.x;
    const int w = t >> 5, lane = t & 31;
    const int g = lane >> 4, lc = lane & 15;
    const int wr = w >> 1, wc = w & 1;                // 4x2 wave grid
    const int rowbase = blockIdx.y * 128;
    const int colbase = blockIdx.x * 128;

    v8f acc[2][4];
#pragma unroll
    for (int mt = 0; mt < 2; ++mt)
#pragma unroll
        for (int nt = 0; nt < 4; ++nt) acc[mt][nt] = vzero8();

    for (int kk = 0; kk < Kd; kk += BK) {
        // Stage A tile 128x64: thread reads 32 contiguous elems of one row
#pragma unroll
        for (int i = 0; i < 32; ++i) {
            int f = t * 32 + i;
            int r = f >> 6, c = f & 63;
            As[r][c] = to_bf16(A[(size_t)(rowbase + r) * Kd + kk + c]);
        }
        // Stage W tile 64x128 transposed into Bs[col][k] (coalesced global)
#pragma unroll
        for (int i = 0; i < 32; ++i) {
            int f = t + 256 * i;
            int k = f >> 7, c = f & 127;
            Bs[c][k] = to_bf16(W[(size_t)(kk + k) * Nd + colbase + c]);
        }
        __syncthreads();

#pragma unroll
        for (int kc = 0; kc < 2; ++kc) {
            v16bf a0 = load_frag_a<72>(&As[0][0], wr * 32 + lc,      kc * 32, g);
            v16bf a1 = load_frag_a<72>(&As[0][0], wr * 32 + 16 + lc, kc * 32, g);
#pragma unroll
            for (int nt = 0; nt < 4; ++nt) {
                v16bf bf = load_frag_b<72>(&Bs[0][0], wc * 64 + nt * 16 + lc,
                                           kc * 32, g);
                acc[0][nt] = wmma_bf16(a0, bf, acc[0][nt]);
                acc[1][nt] = wmma_bf16(a1, bf, acc[1][nt]);
            }
        }
        __syncthreads();
    }

    // Epilogue: C layout VGPR r -> row = r + 8g, lanes 0..15 -> cols
#pragma unroll
    for (int mt = 0; mt < 2; ++mt) {
#pragma unroll
        for (int nt = 0; nt < 4; ++nt) {
            int col = colbase + wc * 64 + nt * 16 + lc;
            float bv = bias[col];
#pragma unroll
            for (int r = 0; r < 8; ++r) {
                int row = rowbase + wr * 32 + mt * 16 + r + 8 * g;
                float v = acc[mt][nt][r] + bv;
                if (OUT_BF16)
                    ((__bf16*)outp)[(size_t)row * Nd + col] = (__bf16)v;
                else
                    ((float*)outp)[(size_t)row * Nd + col] = v;
            }
        }
    }
}

// ---------------------------------------------------------------------------
// Kernel: repack kd/vd row-major [bp][nd] bf16 into attention layouts:
//   kd [b][n][kpos][d]  (d contiguous -> scores B fragments)
//   vdT[b][n][d][kpos]  (k contiguous -> a_v    B fragments)
// grid (1024), block 256.
// ---------------------------------------------------------------------------
__global__ __launch_bounds__(256) void repack_k(const __bf16* __restrict__ kt,
                                                const __bf16* __restrict__ vt,
                                                __bf16* __restrict__ kd,
                                                __bf16* __restrict__ vdT) {
    int idx = blockIdx.x * 256 + threadIdx.x;     // over 256*1024
    int bp = idx >> 10, nd = idx & 1023;
    int b = bp >> 6, p = bp & 63, n = nd >> 6, d = nd & 63;
    kd [(((size_t)(b * 16 + n)) * 64 + p) * 64 + d] = kt[idx];
    vdT[(((size_t)(b * 16 + n)) * 64 + d) * 64 + p] = vt[idx];
}

// ---------------------------------------------------------------------------
// Kernel: attention. grid (S/128, N, B), block 256 = 8 waves, wave = 16 rows.
// scores = q kd^T * 0.125 -> softmax over 64 -> a_v = probs vd  (all WMMA bf16)
// ---------------------------------------------------------------------------
__global__ __launch_bounds__(256) void attn_k(const __bf16* __restrict__ q,
                                              const __bf16* __restrict__ kd,
                                              const __bf16* __restrict__ vdT,
                                              __bf16* __restrict__ av) {
    constexpr int S = 4096, Nh = 16, D = 64, KP = 64;
    __shared__ __bf16 qs [128][72];
    __shared__ __bf16 kds[64][72];
    __shared__ __bf16 vds[64][72];
    __shared__ __bf16 ps [128][72];

    const int t = threadIdx.x;
    const int w = t >> 5, lane = t & 31;
    const int g = lane >> 4, lc = lane & 15;
    const int b = blockIdx.z, n = blockIdx.y;
    const int s0 = blockIdx.x * 128;

    // Stage q tile (bf16 copy from projected q)
#pragma unroll
    for (int i = 0; i < 32; ++i) {
        int f = t + 256 * i;
        int r = f >> 6, d = f & 63;
        qs[r][d] = q[((size_t)(b * S + s0 + r)) * 1024 + n * 64 + d];
    }
    const __bf16* kdn = kd  + ((size_t)(b * Nh + n)) * KP * D;   // [kpos][d]
    const __bf16* vdn = vdT + ((size_t)(b * Nh + n)) * D * KP;   // [d][kpos]
#pragma unroll
    for (int i = 0; i < 16; ++i) {
        int f = t + 256 * i;
        int r = f >> 6, c = f & 63;
        kds[r][c] = kdn[f];
        vds[r][c] = vdn[f];
    }
    __syncthreads();

    const int rw = w * 16;   // wave's row base inside tile

    // scores: 16 rows x 64 kpos = 4 tiles, K=64 over d (2 WMMA each)
    v16bf a0 = load_frag_a<72>(&qs[0][0], rw + lc, 0, g);
    v16bf a1 = load_frag_a<72>(&qs[0][0], rw + lc, 32, g);
    v8f st[4];
#pragma unroll
    for (int kt = 0; kt < 4; ++kt) {
        v16bf b0 = load_frag_b<72>(&kds[0][0], kt * 16 + lc, 0, g);
        v16bf b1 = load_frag_b<72>(&kds[0][0], kt * 16 + lc, 32, g);
        v8f z = wmma_bf16(a0, b0, vzero8());
        st[kt] = wmma_bf16(a1, b1, z);
    }

    // softmax over 64 keys; row M = j + 8g lives in lanes of one 16-lane half,
    // so xor-shuffles with masks 1,2,4,8 reduce exactly over that row.
    const float scale = 0.125f;   // 1/sqrt(64)
#pragma unroll
    for (int j = 0; j < 8; ++j) {
        float m = fmaxf(fmaxf(st[0][j], st[1][j]), fmaxf(st[2][j], st[3][j]));
        m = fmaxf(m, __shfl_xor(m, 1, 32));
        m = fmaxf(m, __shfl_xor(m, 2, 32));
        m = fmaxf(m, __shfl_xor(m, 4, 32));
        m = fmaxf(m, __shfl_xor(m, 8, 32));
        float e[4];
        float s = 0.f;
#pragma unroll
        for (int kt = 0; kt < 4; ++kt) {
            e[kt] = __expf((st[kt][j] - m) * scale);
            s += e[kt];
        }
        s += __shfl_xor(s, 1, 32);
        s += __shfl_xor(s, 2, 32);
        s += __shfl_xor(s, 4, 32);
        s += __shfl_xor(s, 8, 32);
        float rinv = 1.0f / s;
        int row = rw + j + 8 * g;
#pragma unroll
        for (int kt = 0; kt < 4; ++kt)
            ps[row][kt * 16 + lc] = (__bf16)(e[kt] * rinv);
    }
    __syncthreads();

    // a_v = probs (16x64) x vd (64x64): 4 tiles over d, K=64 over kpos
    v16bf pa0 = load_frag_a<72>(&ps[0][0], rw + lc, 0, g);
    v16bf pa1 = load_frag_a<72>(&ps[0][0], rw + lc, 32, g);
#pragma unroll
    for (int dt = 0; dt < 4; ++dt) {
        v16bf b0 = load_frag_b<72>(&vds[0][0], dt * 16 + lc, 0, g);
        v16bf b1 = load_frag_b<72>(&vds[0][0], dt * 16 + lc, 32, g);
        v8f z = wmma_bf16(pa0, b0, vzero8());
        z = wmma_bf16(pa1, b1, z);
        int col = n * 64 + dt * 16 + lc;
#pragma unroll
        for (int r = 0; r < 8; ++r) {
            int row = s0 + rw + r + 8 * g;
            av[((size_t)(b * S + row)) * 1024 + col] = (__bf16)z[r];
        }
    }
}

// ---------------------------------------------------------------------------
extern "C" void kernel_launch(void* const* d_in, const int* in_sizes, int n_in,
                              void* d_out, int out_size, void* d_ws, size_t ws_size,
                              hipStream_t stream) {
    const float* query = (const float*)d_in[0];
    const float* key   = (const float*)d_in[1];
    const float* value = (const float*)d_in[2];
    const float* Wq    = (const float*)d_in[3];
    const float* bq    = (const float*)d_in[4];
    const float* Wk    = (const float*)d_in[5];
    const float* bk    = (const float*)d_in[6];
    const float* Wv    = (const float*)d_in[7];
    const float* bv    = (const float*)d_in[8];
    const float* Wo    = (const float*)d_in[9];
    const float* bo    = (const float*)d_in[10];

    char* ws = (char*)d_ws;
    __bf16* qbuf  = (__bf16*)(ws);                                  // 32 MB
    __bf16* avbuf = (__bf16*)(ws + ((size_t)32 << 20));             // 32 MB
    float*  kp    = (float*) (ws + ((size_t)64 << 20));             // 1 MB
    float*  vp    = (float*) (ws + ((size_t)65 << 20));             // 1 MB
    __bf16* kdb   = (__bf16*)(ws + ((size_t)66 << 20));             // 512 KB
    __bf16* vdTb  = (__bf16*)(ws + ((size_t)66 << 20) + (512u << 10));
    __bf16* ktmp  = (__bf16*)(ws + ((size_t)67 << 20));             // 512 KB
    __bf16* vtmp  = (__bf16*)(ws + ((size_t)67 << 20) + (512u << 10));

    // 1. pool K,V along S (linear => pool before projection)
    pool_k<<<dim3(64, 4, 2), 256, 0, stream>>>(key, value, kp, vp);
    // 2. project pooled K,V (tiny WMMA GEMMs, M=256)
    gemm_bias_k<float, true><<<dim3(8, 2), 256, 0, stream>>>(kp, Wk, bk, ktmp, 256);
    gemm_bias_k<float, true><<<dim3(8, 2), 256, 0, stream>>>(vp, Wv, bv, vtmp, 256);
    // 3. repack into attention fragment layouts
    repack_k<<<dim3(1024), 256, 0, stream>>>(ktmp, vtmp, kdb, vdTb);
    // 4. q projection (M=16384)
    gemm_bias_k<float, true><<<dim3(8, 128), 256, 0, stream>>>(query, Wq, bq, qbuf, 16384);
    // 5. attention
    attn_k<<<dim3(32, 16, 4), 256, 0, stream>>>(qbuf, kdb, vdTb, avbuf);
    // 6. output projection -> f32 d_out
    gemm_bias_k<__bf16, false><<<dim3(8, 128), 256, 0, stream>>>(avbuf, Wo, bo,
                                                                 d_out, 16384);
}
